// BiLSTM_CRF_44435731645126
// MI455X (gfx1250) — matrix operand order repaired and checked by reading, not compiled, not run on hardware
//
#include <hip/hip_runtime.h>
#include <hip/hip_bf16.h>

// ---------------------------------------------------------------------------
// BiLSTM-CRF forward for MI455X (gfx1250, wave32, WMMA).
//   T=2048, E=512, Hh=512/dir, 4*Hh=2048 gates/step, NTAGS=5.
// Pipeline:
//   1) gather_x:       X[t][e] = (bf16) emb[sentence[t]][e]
//   2) pack_w (x4):    repack W (NxK row-major f32) into WMMA B-fragment
//                      order (bf16): frag(nt,kt) is 32 lanes x 16 halves
//                      contiguous.
//   3) input_gemm(x2): Gin = X(rev?) @ W_ih^T via v_wmma_f32_16x16x32_bf16.
//   4) lstm_rec:       grid = 8 (2 dirs x 4 WGs).  Each WG owns 128 hidden
//                      units (= 512 W_hh rows, 512KB bf16/step) so the
//                      recurrent W_hh stream runs on 4 WGPs per direction.
//                      h exchanged through L2 with agent-scope atomics + an
//                      accumulating arrival counter per direction.  A-frags
//                      live in VGPRs (fully unrolled, constant indices).
//   5) feats / 6) crf: tiny projection + serial logsumexp scan.
// ---------------------------------------------------------------------------

typedef __attribute__((ext_vector_type(16))) __bf16 v16bf;
typedef __attribute__((ext_vector_type(8)))  __bf16 v8bf;
typedef __attribute__((ext_vector_type(8)))  float  v8f;

#define TT      2048
#define EE      512
#define HH2     512          // Hh
#define G4      2048         // 4*Hh
#define HFULL   1024
#define NTAGS   5
#define START_T 3
#define STOP_T  4
#define NEGV    (-10000.0f)
#define NWGD    4            // workgroups per direction in recurrence

// ---------------------------------------------------------------- kernel 1
__global__ void gather_x_kernel(const int* __restrict__ sentence,
                                const float* __restrict__ emb,
                                __bf16* __restrict__ Xbf) {
    int idx = blockIdx.x * blockDim.x + threadIdx.x;   // over T*E
    int t = idx / EE;
    int e = idx - t * EE;
    if (t < TT) {
        int tok = sentence[t];
        Xbf[idx] = (__bf16)emb[(size_t)tok * EE + e];
    }
}

// ---------------------------------------------------------------- kernel 2
__global__ void pack_w_kernel(const float* __restrict__ W,
                              __bf16* __restrict__ out,
                              int N, int K) {
    int idx = blockIdx.x * blockDim.x + threadIdx.x;   // over N*K
    if (idx >= N * K) return;
    int frag = idx >> 9;          // /512
    int r    = idx & 511;
    int lane = r >> 4;
    int j    = r & 15;
    int ktiles = K >> 5;          // K/32
    int nt = frag / ktiles;
    int kt = frag - nt * ktiles;
    int n  = nt * 16 + (lane & 15);
    int k  = kt * 32 + ((lane >= 16) ? 8 : 0) + j + ((j >= 8) ? 8 : 0);
    out[idx] = (__bf16)W[(size_t)n * K + k];
}

// ---------------------------------------------------------------- kernel 3
// grid = (T/16, G4/128), block 256 (8 waves), one 16x16 C tile per wave.
__global__ void input_gemm_kernel(const __bf16* __restrict__ X,
                                  const __bf16* __restrict__ Wpk,
                                  float* __restrict__ Gout,
                                  int reverse) {
    const int lane = threadIdx.x & 31;
    const int wave = threadIdx.x >> 5;
    const bool lo  = lane < 16;
    const int mbase = blockIdx.x * 16;
    const int nt    = blockIdx.y * 8 + wave;        // 0..127

    int rowm = mbase + (lane & 15);
    int row  = reverse ? (TT - 1 - rowm) : rowm;
    const __bf16* arow = X + (size_t)row * EE;

    v8f acc;
    #pragma unroll
    for (int i = 0; i < 8; ++i) acc[i] = 0.0f;

    union AV { v16bf v; v8bf h[2]; } a;
    #pragma unroll 4
    for (int kt = 0; kt < EE / 32; ++kt) {
        const __bf16* ap = arow + kt * 32 + (lo ? 0 : 8);
        a.h[0] = *(const v8bf*)(ap);
        a.h[1] = *(const v8bf*)(ap + 16);
        const __bf16* bp = Wpk + ((size_t)nt * (EE / 32) + kt) * 512 + lane * 16;
        v16bf b = *(const v16bf*)bp;
        acc = __builtin_amdgcn_wmma_f32_16x16x32_bf16(
                  false, a.v, false, b, (short)0, acc, false, false);
    }

    const int n     = nt * 16 + (lane & 15);
    const int mrow0 = mbase + (lo ? 0 : 8);
    #pragma unroll
    for (int i = 0; i < 8; ++i)
        Gout[(size_t)(mrow0 + i) * G4 + n] = acc[i];
}

// ---------------------------------------------------------------- kernel 4a
__global__ void init_sync_kernel(int* __restrict__ cnt) {
    if (threadIdx.x < 2) cnt[threadIdx.x] = 0;
}

// ---------------------------------------------------------------- kernel 4
// grid = 8 (dir = blk/4, slice k = blk%4), block 512 (16 waves).
// WG k of a direction owns hidden units [128k,128k+128): gate rows
// n = g*512 + 128k + [0,128) for g=0..3  ->  32 16-wide N tiles.
__global__ void lstm_rec_kernel(const float* __restrict__ GinF,
                                const float* __restrict__ GinB,
                                const __bf16* __restrict__ WhhF,
                                const __bf16* __restrict__ WhhB,
                                const float* __restrict__ bF,
                                const float* __restrict__ bB,
                                const float* __restrict__ h0,
                                const float* __restrict__ c0,
                                float* __restrict__ HS,
                                float* __restrict__ Hpub,   // [2][512]
                                int* __restrict__ cnt) {    // [2]
    const int dir  = blockIdx.x >> 2;
    const int kwg  = blockIdx.x & 3;
    const int tid  = threadIdx.x;
    const int lane = tid & 31;
    const int wave = tid >> 5;
    const bool lo  = lane < 16;

    const float*  Gin  = dir ? GinB : GinF;
    const __bf16* Wpk  = dir ? WhhB : WhhF;
    const float*  bias = dir ? bB : bF;
    float*        hpub = Hpub + dir * HH2;
    int*          ctr  = cnt + dir;

    __shared__ __bf16 hbf[HH2];
    __shared__ float  gl[4 * 128];     // this WG's gate slice

    float c = 0.0f;
    if (tid < 128) c = c0[dir * HH2 + kwg * 128 + tid];

    // persistent A fragments, kept in VGPRs: all indices below are
    // compile-time constants (fully unrolled loops).  Dead lanes stay zero.
    union AV { v16bf v; v8bf h[2]; __bf16 e[16]; } af[HH2 / 32];
    #pragma unroll
    for (int kt = 0; kt < HH2 / 32; ++kt)
        #pragma unroll
        for (int i = 0; i < 16; ++i) af[kt].e[i] = (__bf16)0.0f;

    // wave's two output tiles: pair p = wave, wave+16; p -> (g = p>>3, lt = p&7)
    const int p0 = wave, p1 = wave + 16;

    for (int t = 0; t < TT; ++t) {
        // ---- 1) pull current h (step 0: h0; else published h) into LDS bf16
        if (t == 0) {
            hbf[tid] = (__bf16)h0[dir * HH2 + tid];
        } else {
            float hv = __hip_atomic_load(hpub + tid, __ATOMIC_RELAXED,
                                         __HIP_MEMORY_SCOPE_AGENT);
            hbf[tid] = (__bf16)hv;
        }
        __syncthreads();

        // ---- 2) build A fragments once per step (lanes with M==0 only)
        if ((lane & 15) == 0) {
            #pragma unroll
            for (int kt = 0; kt < HH2 / 32; ++kt) {
                const __bf16* hp = hbf + kt * 32 + (lo ? 0 : 8);
                af[kt].h[0] = *(const v8bf*)(hp);
                af[kt].h[1] = *(const v8bf*)(hp + 16);
            }
        }

        // ---- 3) gate matvec: two 16-wide tiles per wave (fully unrolled)
        #pragma unroll
        for (int pp = 0; pp < 2; ++pp) {
            const int p  = pp ? p1 : p0;
            const int g  = p >> 3;
            const int lt = p & 7;
            const int nt = g * 32 + kwg * 8 + lt;   // global N tile
            const __bf16* bbase = Wpk + (size_t)nt * (HH2 / 32) * 512
                                      + lane * 16;
            v8f acc;
            #pragma unroll
            for (int i = 0; i < 8; ++i) acc[i] = 0.0f;
            #pragma unroll
            for (int kt = 0; kt < HH2 / 32; ++kt) {
                v16bf b = *(const v16bf*)(bbase + kt * 512);
                acc = __builtin_amdgcn_wmma_f32_16x16x32_bf16(
                          false, af[kt].v, false, b, (short)0, acc,
                          false, false);
            }
            if (lane < 16) {                        // row M=0 lives here
                int n = nt * 16 + lane;             // global gate index
                gl[g * 128 + lt * 16 + lane] =
                    acc[0] + Gin[(size_t)t * G4 + n] + bias[n];
            }
        }
        __syncthreads();

        // ---- 4) elementwise cell for this WG's 128 units
        if (tid < 128) {
            int j = kwg * 128 + tid;                // unit within direction
            float gi = gl[tid];
            float gf = gl[128 + tid];
            float gg = gl[256 + tid];
            float go = gl[384 + tid];
            float si = 1.0f / (1.0f + __expf(-gi));
            float sf = 1.0f / (1.0f + __expf(-gf));
            float so = 1.0f / (1.0f + __expf(-go));
            c = sf * c + si * tanhf(gg);
            float h = so * tanhf(c);
            int outrow = dir ? (TT - 1 - t) : t;
            HS[(size_t)outrow * HFULL + dir * HH2 + j] = h;
            __hip_atomic_store(hpub + j, h, __ATOMIC_RELAXED,
                               __HIP_MEMORY_SCOPE_AGENT);
        }
        __syncthreads();

        // ---- 5) cross-WG step barrier (skip after last step)
        if (t + 1 < TT) {
            if (tid == 0) {
                __hip_atomic_fetch_add(ctr, 1, __ATOMIC_RELEASE,
                                       __HIP_MEMORY_SCOPE_AGENT);
                int target = NWGD * (t + 1);
                while (__hip_atomic_load(ctr, __ATOMIC_ACQUIRE,
                                         __HIP_MEMORY_SCOPE_AGENT) < target)
                    __builtin_amdgcn_s_sleep(1);
            }
            __syncthreads();
        }
    }
}

// ---------------------------------------------------------------- kernel 5
__global__ void feats_kernel(const float* __restrict__ HS,
                             const float* __restrict__ w_out,
                             const float* __restrict__ b_out,
                             float* __restrict__ feats) {
    int idx = blockIdx.x * blockDim.x + threadIdx.x;   // over T*NTAGS
    if (idx >= TT * NTAGS) return;
    int t   = idx / NTAGS;
    int tag = idx - t * NTAGS;
    const float4* hp = (const float4*)(HS + (size_t)t * HFULL);
    const float4* wp = (const float4*)(w_out + (size_t)tag * HFULL);
    float acc = 0.0f;
    #pragma unroll 4
    for (int k = 0; k < HFULL / 4; ++k) {
        float4 hv = hp[k];
        float4 wv = wp[k];
        acc += hv.x * wv.x + hv.y * wv.y + hv.z * wv.z + hv.w * wv.w;
    }
    feats[idx] = acc + b_out[tag];
}

// ---------------------------------------------------------------- kernel 6
__global__ void crf_kernel(const float* __restrict__ feats,
                           const float* __restrict__ trans,
                           float* __restrict__ out) {
    if (threadIdx.x != 0 || blockIdx.x != 0) return;
    float fv[NTAGS];
    for (int i = 0; i < NTAGS; ++i) fv[i] = NEGV;
    fv[START_T] = 0.0f;
    for (int t = 0; t < TT; ++t) {
        float nv[NTAGS];
        for (int j = 0; j < NTAGS; ++j) {
            float m = NEGV * 4.0f;
            float v[NTAGS];
            for (int i = 0; i < NTAGS; ++i) {
                v[i] = fv[i] + trans[j * NTAGS + i];
                m = fmaxf(m, v[i]);
            }
            float s = 0.0f;
            for (int i = 0; i < NTAGS; ++i) s += __expf(v[i] - m);
            nv[j] = __logf(s) + m + feats[t * NTAGS + j];
        }
        for (int j = 0; j < NTAGS; ++j) fv[j] = nv[j];
    }
    float m = NEGV * 4.0f;
    float v[NTAGS];
    for (int i = 0; i < NTAGS; ++i) {
        v[i] = fv[i] + trans[STOP_T * NTAGS + i];
        m = fmaxf(m, v[i]);
    }
    float s = 0.0f;
    for (int i = 0; i < NTAGS; ++i) s += __expf(v[i] - m);
    out[0] = __logf(s) + m;
}

// ---------------------------------------------------------------- launcher
extern "C" void kernel_launch(void* const* d_in, const int* in_sizes, int n_in,
                              void* d_out, int out_size, void* d_ws, size_t ws_size,
                              hipStream_t stream) {
    const int*   sentence = (const int*)  d_in[0];
    const float* emb      = (const float*)d_in[1];
    const float* w_ih_f   = (const float*)d_in[2];
    const float* w_hh_f   = (const float*)d_in[3];
    const float* b_f      = (const float*)d_in[4];
    const float* w_ih_b   = (const float*)d_in[5];
    const float* w_hh_b   = (const float*)d_in[6];
    const float* b_b      = (const float*)d_in[7];
    const float* w_out    = (const float*)d_in[8];
    const float* b_out    = (const float*)d_in[9];
    const float* trans    = (const float*)d_in[10];
    const float* h0       = (const float*)d_in[11];
    const float* c0       = (const float*)d_in[12];
    float* outp = (float*)d_out;

    // workspace carve-up (all 256B aligned)
    char* ws = (char*)d_ws;
    __bf16* Xbf   = (__bf16*)(ws);                       // 2 MB
    __bf16* WihFp = (__bf16*)(ws + (2u  << 20));         // 2 MB
    __bf16* WihBp = (__bf16*)(ws + (4u  << 20));         // 2 MB
    __bf16* WhhFp = (__bf16*)(ws + (6u  << 20));         // 2 MB
    __bf16* WhhBp = (__bf16*)(ws + (8u  << 20));         // 2 MB
    float*  GinF  = (float*) (ws + (10u << 20));         // 16 MB
    float*  GinB  = (float*) (ws + (26u << 20));         // 16 MB
    float*  HS    = (float*) (ws + (42u << 20));         // 8 MB
    float*  feats = (float*) (ws + (50u << 20));         // 40 KB
    float*  Hpub  = (float*) (ws + (51u << 20));         // 4 KB
    int*    cnt   = (int*)   (ws + (51u << 20) + 8192);  // 8 B

    // 1) gather + bf16 convert
    gather_x_kernel<<<(TT * EE) / 256, 256, 0, stream>>>(sentence, emb, Xbf);

    // 2) pack all four weight matrices into WMMA B-fragment order
    const int NW = G4 * EE;                              // == G4*HH2 too
    pack_w_kernel<<<NW / 256, 256, 0, stream>>>(w_ih_f, WihFp, G4, EE);
    pack_w_kernel<<<NW / 256, 256, 0, stream>>>(w_ih_b, WihBp, G4, EE);
    pack_w_kernel<<<NW / 256, 256, 0, stream>>>(w_hh_f, WhhFp, G4, HH2);
    pack_w_kernel<<<NW / 256, 256, 0, stream>>>(w_hh_b, WhhBp, G4, HH2);

    // 3) input-side GEMMs (fwd uses X, bwd uses row-reversed X)
    dim3 ggrid(TT / 16, G4 / 128);
    input_gemm_kernel<<<ggrid, 256, 0, stream>>>(Xbf, WihFp, GinF, 0);
    input_gemm_kernel<<<ggrid, 256, 0, stream>>>(Xbf, WihBp, GinB, 1);

    // 4) recurrences: 4 WGPs per direction, L2-mediated h exchange
    init_sync_kernel<<<1, 32, 0, stream>>>(cnt);
    lstm_rec_kernel<<<2 * NWGD, 512, 0, stream>>>(GinF, GinB, WhhFp, WhhBp,
                                                  b_f, b_b, h0, c0, HS,
                                                  Hpub, cnt);

    // 5) tag projection
    feats_kernel<<<(TT * NTAGS + 255) / 256, 256, 0, stream>>>(HS, w_out,
                                                               b_out, feats);
    // 6) CRF forward scan -> scalar
    crf_kernel<<<1, 32, 0, stream>>>(feats, trans, outp);
}
